// BlockGNNComposer_10806137716788
// MI455X (gfx1250) — matrix operand order repaired and checked by reading, not compiled, not run on hardware
//
#include <hip/hip_runtime.h>

#define K_BLK  20000
#define EB_N   320000
#define E2_N   640000
#define NCV_N  100000
#define DIN_P  448          // 400-wide MLP input zero-padded to a multiple of 64

typedef __attribute__((ext_vector_type(16))) __bf16 v16bf;
typedef __attribute__((ext_vector_type(8)))  float  v8f;

union BfVec { v16bf v; unsigned int u[8]; uint4 q[2]; };

__device__ __forceinline__ unsigned short f2bf(float f) {
  unsigned int u = __float_as_uint(f);
  u += 0x7FFFu + ((u >> 16) & 1u);          // round-to-nearest-even
  return (unsigned short)(u >> 16);
}
__device__ __forceinline__ float bf2f(unsigned short h) {
  return __uint_as_float(((unsigned int)h) << 16);
}
__device__ __forceinline__ void atomicMaxF(float* addr, float val) {
  if (val >= 0.f) atomicMax((int*)addr, __float_as_int(val));
  else            atomicMin((unsigned int*)addr, __float_as_uint(val));
}

// Load one k-step of fragments: A (16x32) + B (32x16)x4 as contiguous 16B runs.
//   A: lane m=l&15, half=l>>4 -> bytes [k0+8h, +16) and [k0+16+8h, +16)
//   B: lane n=l&15, half=l>>4 -> bytes [k0+8h, +16) and [k0+16+8h, +16)
__device__ __forceinline__ void load_frag(const unsigned short* __restrict__ arow,
                                          const unsigned short* __restrict__ bp0,
                                          long long bstride, int k0, int hf,
                                          BfVec& a, BfVec b[4])
{
  a.q[0] = *(const uint4*)(arow + k0 + 8 * hf);
  a.q[1] = *(const uint4*)(arow + k0 + 16 + 8 * hf);
#pragma unroll
  for (int cc = 0; cc < 4; ++cc) {
    const unsigned short* bp = bp0 + bstride * cc;
    b[cc].q[0] = *(const uint4*)(bp + k0 + 8 * hf);
    b[cc].q[1] = *(const uint4*)(bp + k0 + 16 + 8 * hf);
  }
}

__device__ __forceinline__ void mma4(const BfVec& a, const BfVec b[4], v8f acc[4])
{
#pragma unroll
  for (int cc = 0; cc < 4; ++cc)
    acc[cc] = __builtin_amdgcn_wmma_f32_16x16x32_bf16(false, a.v, false, b[cc].v,
                                                      (short)0, acc[cc], false, false);
}

// ------------------------------------------------------------------
// GEMM: Out[M,N] = act(A[M,Kd](bf16) @ W[Kd,N] + bias),  Kd % 64 == 0.
// Wt pre-packed transposed bf16: Wt[n*Kd + k].
// One wave32 computes a 16x64 strip; double-buffered (ping-pong) fragment
// loads overlap global_load_b128 latency with the 4-WMMA groups.
// D layout: elem r -> row = r + 8*half, col = l&15.
// Dual outputs: fp32 (LN/residual consumers) and bf16 (next GEMM's A).
// ------------------------------------------------------------------
__global__ __launch_bounds__(256) void gemm_wmma(
    const unsigned short* __restrict__ A, const unsigned short* __restrict__ Wt,
    const float* __restrict__ bias, float* __restrict__ OutF,
    unsigned short* __restrict__ OutH, int M, int Kd, int N, int relu)
{
  const int lane   = threadIdx.x & 31;
  const int wv     = threadIdx.x >> 5;
  const int strips = N >> 6;                       // 64-wide strips
  const long long tile  = (long long)blockIdx.x * 8 + wv;
  const long long total = (long long)(M >> 4) * strips;
  if (tile >= total) return;                       // wave-uniform; EXEC stays all-ones
  const int row0 = (int)(tile / strips) * 16;
  const int col0 = (int)(tile % strips) * 64;
  const int hf  = lane >> 4;
  const int l15 = lane & 15;

  const unsigned short* arow = A + (long long)(row0 + l15) * Kd;
  const unsigned short* bp0  = Wt + (long long)(col0 + l15) * Kd;
  const long long bs = (long long)16 * Kd;

  v8f acc[4];
#pragma unroll
  for (int cc = 0; cc < 4; ++cc) acc[cc] = (v8f){0.f,0.f,0.f,0.f,0.f,0.f,0.f,0.f};

  BfVec aA, aB;
  BfVec bA[4], bB[4];
  load_frag(arow, bp0, bs, 0, hf, aA, bA);
  for (int k0 = 0; k0 < Kd; k0 += 64) {
    load_frag(arow, bp0, bs, k0 + 32, hf, aB, bB);   // prefetch odd step
    mma4(aA, bA, acc);
    if (k0 + 64 < Kd) load_frag(arow, bp0, bs, k0 + 64, hf, aA, bA); // prefetch next even
    mma4(aB, bB, acc);
  }

#pragma unroll
  for (int cc = 0; cc < 4; ++cc) {
    const int col = col0 + cc * 16 + l15;
    const float bn = bias ? bias[col] : 0.f;
#pragma unroll
    for (int r = 0; r < 8; ++r) {
      float val = acc[cc][r] + bn;
      if (relu) val = fmaxf(val, 0.f);
      const long long o = (long long)(row0 + r + 8 * hf) * N + col;
      if (OutF) OutF[o] = val;
      if (OutH) OutH[o] = f2bf(val);
    }
  }
}

// ---------------- LayerNorm (wave per row), optional residual + column bias + relu
// Dual outputs: fp32 (nullable) and bf16 mirror (nullable) for downstream GEMM A.
template <int D>
__global__ __launch_bounds__(256) void ln_kernel(
    const float* __restrict__ in, const float* __restrict__ res,
    const float* __restrict__ cb, const float* __restrict__ g,
    const float* __restrict__ b, float* __restrict__ out,
    unsigned short* __restrict__ outh, int M, int relu)
{
  constexpr int PER = D / 32;
  const int lane = threadIdx.x & 31;
  const int row  = blockIdx.x * (blockDim.x >> 5) + (threadIdx.x >> 5);
  if (row >= M) return;
  const float* ip = in + (long long)row * D;
  const float* rp = res ? res + (long long)row * D : nullptr;
  float v[PER];
  float s = 0.f;
#pragma unroll
  for (int i = 0; i < PER; ++i) {
    const int c = lane + 32 * i;
    float t = ip[c];
    if (rp) t += rp[c];
    if (cb) t += cb[c];
    v[i] = t; s += t;
  }
  for (int m = 16; m >= 1; m >>= 1) s += __shfl_xor(s, m, 32);
  const float mean = s / (float)D;
  float s2 = 0.f;
#pragma unroll
  for (int i = 0; i < PER; ++i) { const float d = v[i] - mean; s2 += d * d; }
  for (int m = 16; m >= 1; m >>= 1) s2 += __shfl_xor(s2, m, 32);
  const float inv = rsqrtf(s2 / (float)D + 1e-5f);
  float*          op  = out  ? out  + (long long)row * D : nullptr;
  unsigned short* oph = outh ? outh + (long long)row * D : nullptr;
#pragma unroll
  for (int i = 0; i < PER; ++i) {
    const int c = lane + 32 * i;
    float val = (v[i] - mean) * inv * g[c] + b[c];
    if (relu) val = fmaxf(val, 0.f);
    if (op)  op[c]  = val;
    if (oph) oph[c] = f2bf(val);
  }
}

// ---------------- e = relu(ea[EB,4] @ We_proj[4,256] + be) -> bf16 [EB,256]
__global__ __launch_bounds__(256) void eproj_kernel(
    const float* __restrict__ ea, const float* __restrict__ Wp,
    const float* __restrict__ be, unsigned short* __restrict__ e)
{
  const long long tid = (long long)blockIdx.x * blockDim.x + threadIdx.x;
  if (tid >= (long long)EB_N * 64) return;
  const int row = (int)(tid >> 6);
  const int j0  = ((int)tid & 63) * 4;
  const float a0 = ea[row * 4 + 0], a1 = ea[row * 4 + 1];
  const float a2 = ea[row * 4 + 2], a3 = ea[row * 4 + 3];
#pragma unroll
  for (int j = 0; j < 4; ++j) {
    const int col = j0 + j;
    float v = a0 * Wp[col] + a1 * Wp[256 + col] + a2 * Wp[512 + col] + a3 * Wp[768 + col] + be[col];
    v = fmaxf(v, 0.f);
    e[(long long)row * 256 + col] = f2bf(v);
  }
}

__global__ __launch_bounds__(256) void fill_kernel(float* __restrict__ p, float v, long long n)
{
  const long long tid = (long long)blockIdx.x * blockDim.x + threadIdx.x;
  if (tid < n) p[tid] = v;
}

// ---------------- GATv2 attention logits + segment max (wave per directed edge)
__global__ __launch_bounds__(256) void edge_score_kernel(
    const int* __restrict__ ei0, const int* __restrict__ ei1,
    const float* __restrict__ xl, const float* __restrict__ xr,
    const unsigned short* __restrict__ ee, const float* __restrict__ att,
    float* __restrict__ aex, float* __restrict__ amax)
{
  const int lane = threadIdx.x & 31;
  const long long ew = (long long)blockIdx.x * 8 + (threadIdx.x >> 5);
  if (ew >= (long long)E2_N) return;
  const int eb  = (int)(ew < EB_N ? ew : ew - EB_N);
  const int src = (ew < EB_N) ? ei0[eb] : ei1[eb];
  const int dst = (ew < EB_N) ? ei1[eb] : ei0[eb];
  const float* xls = xl + (long long)src * 256;
  const float* xrd = xr + (long long)dst * 256;
  const unsigned short* eep = ee + (long long)eb * 256;
  float sh[4] = {0.f, 0.f, 0.f, 0.f};
#pragma unroll
  for (int j = 0; j < 8; ++j) {
    const int c = j * 32 + lane;                 // coalesced
    float m = xls[c] + xrd[c] + bf2f(eep[c]);
    m = (m > 0.f) ? m : 0.2f * m;                // leaky_relu(0.2)
    sh[j >> 1] += m * att[(j >> 1) * 64 + ((j & 1) * 32 + lane)];
  }
#pragma unroll
  for (int h = 0; h < 4; ++h) {
    float s = sh[h];
    for (int m = 16; m >= 1; m >>= 1) s += __shfl_xor(s, m, 32);
    sh[h] = s;
  }
  if (lane < 4) {
    const float s = (lane == 0) ? sh[0] : (lane == 1) ? sh[1] : (lane == 2) ? sh[2] : sh[3];
    aex[ew * 4 + lane] = s;
    atomicMaxF(&amax[(long long)dst * 4 + lane], s);
  }
}

// ---------------- ex = exp(a - amax[dst]); den[dst] += ex  (thread per edge*head)
__global__ __launch_bounds__(256) void edge_exp_kernel(
    const int* __restrict__ ei0, const int* __restrict__ ei1,
    float* __restrict__ aex, const float* __restrict__ amax, float* __restrict__ den)
{
  const long long tid = (long long)blockIdx.x * blockDim.x + threadIdx.x;
  if (tid >= (long long)E2_N * 4) return;
  const long long ew = tid >> 2;
  const int h = (int)(tid & 3);
  const int eb  = (int)(ew < EB_N ? ew : ew - EB_N);
  const int dst = (ew < EB_N) ? ei1[eb] : ei0[eb];
  const float ex = __expf(aex[tid] - amax[(long long)dst * 4 + h]);
  aex[tid] = ex;
  atomicAdd(&den[(long long)dst * 4 + h], ex);
}

// ---------------- agg[dst] += xl[src] * alpha  (wave per directed edge)
__global__ __launch_bounds__(256) void edge_agg_kernel(
    const int* __restrict__ ei0, const int* __restrict__ ei1,
    const float* __restrict__ xl, const float* __restrict__ aex,
    const float* __restrict__ den, float* __restrict__ agg)
{
  const int lane = threadIdx.x & 31;
  const long long ew = (long long)blockIdx.x * 8 + (threadIdx.x >> 5);
  if (ew >= (long long)E2_N) return;
  const int eb  = (int)(ew < EB_N ? ew : ew - EB_N);
  const int src = (ew < EB_N) ? ei0[eb] : ei1[eb];
  const int dst = (ew < EB_N) ? ei1[eb] : ei0[eb];
  float al[4];
#pragma unroll
  for (int h = 0; h < 4; ++h)
    al[h] = aex[ew * 4 + h] / (den[(long long)dst * 4 + h] + 1e-16f);
  const float* xls = xl + (long long)src * 256;
  float* ad = agg + (long long)dst * 256;
#pragma unroll
  for (int j = 0; j < 8; ++j) {
    const int c = j * 32 + lane;
    atomicAdd(&ad[c], xls[c] * al[j >> 1]);
  }
}

// ---------------- build MLP input [N,448] bf16: [sub(128)|z[bid](256)|bnd(16)|pad(48)]
__global__ __launch_bounds__(256) void gather_kernel(
    const float* __restrict__ sub, const unsigned short* __restrict__ zb,
    const int* __restrict__ bid, const float* __restrict__ bnd,
    unsigned short* __restrict__ out, int Nrows)
{
  const long long tid = (long long)blockIdx.x * blockDim.x + threadIdx.x;
  if (tid >= (long long)Nrows * DIN_P) return;
  const int row = (int)(tid / DIN_P);
  const int col = (int)(tid % DIN_P);
  unsigned short o;
  if      (col < 128) o = f2bf(sub[(long long)row * 128 + col]);
  else if (col < 384) o = zb[(long long)bid[row] * 256 + (col - 128)];   // already bf16
  else if (col < 400) o = f2bf(bnd[(long long)row * 16 + (col - 384)]);
  else                o = 0;
  out[tid] = o;
}

// ---------------- pack W[Kin,N] fp32 -> Wt[N,Kpad] bf16 (zero-padded K)
__global__ __launch_bounds__(256) void pack_kernel(
    const float* __restrict__ W, unsigned short* __restrict__ Wt,
    int Kin, int Kpad, int N)
{
  const long long tid = (long long)blockIdx.x * blockDim.x + threadIdx.x;
  if (tid >= (long long)N * Kpad) return;
  const int n = (int)(tid / Kpad);
  const int k = (int)(tid % Kpad);
  Wt[tid] = (k < Kin) ? f2bf(W[(long long)k * N + n]) : (unsigned short)0;
}

// ==================================================================
extern "C" void kernel_launch(void* const* d_in, const int* in_sizes, int n_in,
                              void* d_out, int out_size, void* d_ws, size_t ws_size,
                              hipStream_t stream)
{
  (void)n_in; (void)out_size; (void)ws_size;
  const float* bf   = (const float*)d_in[0];
  const int*   ei0  = (const int*)d_in[1];
  const int*   ei1  = ei0 + EB_N;
  const float* ea   = (const float*)d_in[2];
  const int*   cbid = (const int*)d_in[3];
  const int*   vbid = (const int*)d_in[4];
  const float* csub = (const float*)d_in[5];
  const float* vsub = (const float*)d_in[6];
  const float* cbnd = (const float*)d_in[7];
  const float* vbnd = (const float*)d_in[8];

  // Detect param-leaf flattening order: insertion (ln_in_g, 512) vs jax-sorted (We_proj, 1024).
  const bool sorted = (in_sizes[9] == 4 * 256);
  int iWin, ibin, iWep, ibep, ilng, ilnb, iWo1, ibo1, iWo2, ibo2;
  int iWl[2], ibl[2], iWr[2], ibr[2], iWe[2], iatt[2], ibias[2], ing[2], inb[2];
  int iW1[2], ib1[2], ig1[2], ibe1[2], iW2[2], ib2[2], ig2[2], ibe2[2], iW3[2], ib3[2]; // [0]=cons,[1]=var
  if (!sorted) {
    ilng = 9; ilnb = 10; iWin = 11; ibin = 12; iWep = 13; ibep = 14;
    int base = 15;
    for (int c = 0; c < 2; ++c) {
      iWl[c] = base; ibl[c] = base + 1; iWr[c] = base + 2; ibr[c] = base + 3;
      iWe[c] = base + 4; iatt[c] = base + 5; ibias[c] = base + 6; base += 7;
    }
    ing[0] = 29; inb[0] = 30; ing[1] = 31; inb[1] = 32;
    iWo1 = 33; ibo1 = 34; iWo2 = 35; ibo2 = 36;
    const int mb[2] = {47, 37};  // cons_mlp after var_mlp in insertion order
    for (int m = 0; m < 2; ++m) {
      const int b = mb[m];
      iW1[m] = b; ib1[m] = b + 1; ig1[m] = b + 2; ibe1[m] = b + 3; iW2[m] = b + 4;
      ib2[m] = b + 5; ig2[m] = b + 6; ibe2[m] = b + 7; iW3[m] = b + 8; ib3[m] = b + 9;
    }
  } else {
    iWep = 9; iWin = 10; iWo1 = 11; iWo2 = 12; ibep = 13; ibin = 14; ibo1 = 15; ibo2 = 16;
    const int mb[2] = {17, 47};  // cons_mlp sorts before conv*, var_mlp last
    for (int m = 0; m < 2; ++m) {
      const int b = mb[m];
      iW1[m] = b; iW2[m] = b + 1; iW3[m] = b + 2; ib1[m] = b + 3; ib2[m] = b + 4;
      ib3[m] = b + 5; ibe1[m] = b + 6; ibe2[m] = b + 7; ig1[m] = b + 8; ig2[m] = b + 9;
    }
    int base = 27;
    for (int c = 0; c < 2; ++c) {
      iWe[c] = base; iWl[c] = base + 1; iWr[c] = base + 2; iatt[c] = base + 3;
      ibias[c] = base + 4; ibl[c] = base + 5; ibr[c] = base + 6; base += 7;
    }
    ilnb = 41; ilng = 42; inb[0] = 43; ing[0] = 44; inb[1] = 45; ing[1] = 46;
  }
  auto P = [&](int i) -> const float* { return (const float*)d_in[i]; };

  // -------- workspace layout --------
  char* wp = (char*)d_ws;
  auto take = [&](size_t bytes) -> void* {
    void* r = (void*)wp; wp += (bytes + 255) & ~(size_t)255; return r;
  };
  float* x    = (float*)take((size_t)K_BLK * 256 * 4);
  float* xl   = (float*)take((size_t)K_BLK * 256 * 4);
  float* xr   = (float*)take((size_t)K_BLK * 256 * 4);
  float* agg  = (float*)take((size_t)K_BLK * 256 * 4);
  unsigned short* xb = (unsigned short*)take((size_t)K_BLK * 256 * 2);  // bf16 mirror of x
  float* amax = (float*)take((size_t)K_BLK * 4 * 4);
  float* den  = (float*)take((size_t)K_BLK * 4 * 4);
  float* aex  = (float*)take((size_t)E2_N * 4 * 4);
  unsigned short* ebuf  = (unsigned short*)take((size_t)EB_N * 256 * 2); // e bf16; later t1F+h1b
  unsigned short* eebuf = (unsigned short*)take((size_t)EB_N * 256 * 2); // ee bf16; later in448+h2b
  float* bufC = (float*)take((size_t)NCV_N * 256 * 4); // LN512 bf16 temp; later t2F
  unsigned short* WinT = (unsigned short*)take((size_t)256 * 512 * 2);
  unsigned short* WlT[2]; unsigned short* WrT[2]; unsigned short* WeT[2];
  for (int c = 0; c < 2; ++c) {
    WlT[c] = (unsigned short*)take((size_t)256 * 256 * 2);
    WrT[c] = (unsigned short*)take((size_t)256 * 256 * 2);
    WeT[c] = (unsigned short*)take((size_t)256 * 256 * 2);
  }
  unsigned short* Wo1T = (unsigned short*)take((size_t)256 * 256 * 2);
  unsigned short* Wo2T = (unsigned short*)take((size_t)256 * 256 * 2);
  unsigned short* W1T[2]; unsigned short* W2T[2]; unsigned short* W3T[2];
  for (int m = 0; m < 2; ++m) {
    W1T[m] = (unsigned short*)take((size_t)256 * DIN_P * 2);
    W2T[m] = (unsigned short*)take((size_t)256 * 256 * 2);
    W3T[m] = (unsigned short*)take((size_t)128 * 256 * 2);
  }

  auto cdiv = [](long long a, long long b) -> int { return (int)((a + b - 1) / b); };
  auto pack = [&](const float* W, unsigned short* Wt, int Kin, int Kpad, int N) {
    pack_kernel<<<cdiv((long long)N * Kpad, 256), 256, 0, stream>>>(W, Wt, Kin, Kpad, N);
  };
  auto gemm = [&](const unsigned short* A, const unsigned short* Wt, const float* b,
                  float* OF, unsigned short* OH, int M, int Kd, int N, int relu) {
    gemm_wmma<<<cdiv((long long)(M / 16) * (N / 64), 8), 256, 0, stream>>>(
        A, Wt, b, OF, OH, M, Kd, N, relu);
  };

  // -------- pack all weights to transposed bf16 --------
  pack(P(iWin), WinT, 512, 512, 256);
  for (int c = 0; c < 2; ++c) {
    pack(P(iWl[c]), WlT[c], 256, 256, 256);
    pack(P(iWr[c]), WrT[c], 256, 256, 256);
    pack(P(iWe[c]), WeT[c], 256, 256, 256);
  }
  pack(P(iWo1), Wo1T, 256, 256, 256);
  pack(P(iWo2), Wo2T, 256, 256, 256);
  for (int m = 0; m < 2; ++m) {
    pack(P(iW1[m]), W1T[m], 400, DIN_P, 256);
    pack(P(iW2[m]), W2T[m], 256, 256, 256);
    pack(P(iW3[m]), W3T[m], 256, 256, 128);
  }

  // -------- block GNN --------
  unsigned short* ln512b = (unsigned short*)bufC;   // bf16 LN(block_features)
  ln_kernel<512><<<cdiv(K_BLK, 8), 256, 0, stream>>>(bf, nullptr, nullptr, P(ilng), P(ilnb),
                                                     nullptr, ln512b, K_BLK, 0);
  gemm(ln512b, WinT, P(ibin), x, xb, K_BLK, 512, 256, 1);
  eproj_kernel<<<cdiv((long long)EB_N * 64, 256), 256, 0, stream>>>(ea, P(iWep), P(ibep), ebuf);

  for (int c = 0; c < 2; ++c) {
    gemm(xb, WlT[c], P(ibl[c]), xl, nullptr, K_BLK, 256, 256, 0);
    gemm(xb, WrT[c], P(ibr[c]), xr, nullptr, K_BLK, 256, 256, 0);
    gemm(ebuf, WeT[c], nullptr, nullptr, eebuf, EB_N, 256, 256, 0);
    fill_kernel<<<cdiv((long long)K_BLK * 4, 256), 256, 0, stream>>>(amax, -3.402823466e38f,
                                                                     (long long)K_BLK * 4);
    fill_kernel<<<cdiv((long long)K_BLK * 4, 256), 256, 0, stream>>>(den, 0.f, (long long)K_BLK * 4);
    fill_kernel<<<cdiv((long long)K_BLK * 256, 256), 256, 0, stream>>>(agg, 0.f,
                                                                       (long long)K_BLK * 256);
    edge_score_kernel<<<cdiv((long long)E2_N, 8), 256, 0, stream>>>(ei0, ei1, xl, xr, eebuf,
                                                                    P(iatt[c]), aex, amax);
    edge_exp_kernel<<<cdiv((long long)E2_N * 4, 256), 256, 0, stream>>>(ei0, ei1, aex, amax, den);
    edge_agg_kernel<<<cdiv((long long)E2_N, 8), 256, 0, stream>>>(ei0, ei1, xl, aex, den, agg);
    ln_kernel<256><<<cdiv(K_BLK, 8), 256, 0, stream>>>(x, agg, P(ibias[c]), P(ing[c]), P(inb[c]),
                                                       x, xb, K_BLK, 0);
  }
  // z = relu(x@Wo1+bo1)@Wo2+bo2  (bf16 z lands in zb)
  unsigned short* tb = (unsigned short*)xl;
  unsigned short* zb = (unsigned short*)xr;
  gemm(xb, Wo1T, P(ibo1), nullptr, tb, K_BLK, 256, 256, 1);
  gemm(tb, Wo2T, P(ibo2), nullptr, zb, K_BLK, 256, 256, 0);

  // -------- per-node MLP heads: m=0 cons -> d_out[0..], m=1 vars -> d_out[NC*128..]
  float*          t1F   = (float*)ebuf;                                             // 102.4 MB
  unsigned short* h1b   = (unsigned short*)((char*)ebuf + (size_t)NCV_N * 256 * 4); // 51.2 MB
  unsigned short* in448 = eebuf;                                                    // 89.6 MB
  unsigned short* h2b   = (unsigned short*)((char*)eebuf + (size_t)NCV_N * DIN_P * 2); // 51.2 MB
  float*          t2F   = bufC;                                                     // 102.4 MB
  const float* subp[2] = {csub, vsub};
  const int*   bidp[2] = {cbid, vbid};
  const float* bndp[2] = {cbnd, vbnd};
  for (int m = 0; m < 2; ++m) {
    float* outp = (float*)d_out + (size_t)m * (size_t)NCV_N * 128;
    gather_kernel<<<cdiv((long long)NCV_N * DIN_P, 256), 256, 0, stream>>>(subp[m], zb, bidp[m],
                                                                           bndp[m], in448, NCV_N);
    gemm(in448, W1T[m], P(ib1[m]), t1F, nullptr, NCV_N, DIN_P, 256, 0);
    ln_kernel<256><<<cdiv(NCV_N, 8), 256, 0, stream>>>(t1F, nullptr, nullptr, P(ig1[m]),
                                                       P(ibe1[m]), nullptr, h1b, NCV_N, 1);
    gemm(h1b, W2T[m], P(ib2[m]), t2F, nullptr, NCV_N, 256, 256, 0);
    ln_kernel<256><<<cdiv(NCV_N, 8), 256, 0, stream>>>(t2F, nullptr, nullptr, P(ig2[m]),
                                                       P(ibe2[m]), nullptr, h2b, NCV_N, 1);
    gemm(h2b, W3T[m], P(ib3[m]), outp, nullptr, NCV_N, 256, 128, 0);
  }
}
